// TransformerModel_2310692405800
// MI455X (gfx1250) — matrix operand order repaired and checked by reading, not compiled, not run on hardware
//
#include <hip/hip_runtime.h>
#include <hip/hip_bf16.h>

// ---------------- model constants ----------------
#define Dm    512
#define Hh    8
#define HDd   64
#define FFf   2048
#define Ll    6
#define Vv    10000
#define VPAD  10112          // ceil(10000/128)*128
#define Bb    8
#define Ss    512
#define Tt    512
#define Mrows (Bb * Ss)      // 4096 (== Bb*Tt)
#define DDsz  (Dm * Dm)      // 262144
#define DFFsz (Dm * FFf)     // 1048576

typedef __attribute__((ext_vector_type(16))) _Float16 v16h;
typedef __attribute__((ext_vector_type(8)))  _Float16 v8h;
typedef __attribute__((ext_vector_type(8)))  float    v8f;

union AFrag { v16h v; v8h h[2]; };

static __device__ __forceinline__ v8f wmma_acc(const AFrag& a, const AFrag& b, v8f c) {
  // D = A(16x32 f16) * B(32x16 f16) + C(16x16 f32)
  return __builtin_amdgcn_wmma_f32_16x16x32_f16(false, a.v, false, b.v,
                                                (short)0, c, false, false);
}

// ---- CDNA5 async global->LDS copy (ASYNCcnt) with safe fallback ----
#if defined(__has_builtin)
#if __has_builtin(__builtin_amdgcn_global_load_async_to_lds_b128) && \
    __has_builtin(__builtin_amdgcn_s_wait_asynccnt)
#define USE_ASYNC_LDS 1
#endif
#endif
#ifndef USE_ASYNC_LDS
#define USE_ASYNC_LDS 0
#endif

#if USE_ASYNC_LDS
typedef int v4i __attribute__((vector_size(16)));
typedef __attribute__((address_space(1))) v4i* as1_v4i;   // global (prints as __device__)
typedef __attribute__((address_space(3))) v4i* as3_v4i;   // LDS    (prints as __shared__)
static __device__ __forceinline__ void async_cp16(const void* g, void* l) {
  __builtin_amdgcn_global_load_async_to_lds_b128((as1_v4i)g, (as3_v4i)l, 0, 0);
}
static __device__ __forceinline__ void async_drain() {
  __builtin_amdgcn_s_wait_asynccnt(0);
}
#else
static __device__ __forceinline__ void async_cp16(const void* g, void* l) {
  *(v8h*)l = *(const v8h*)g;          // global_load_b128 + ds_store_b128
}
static __device__ __forceinline__ void async_drain() {}
#endif

// ---------------------------------------------------------------------------
// Weight prep: f32 [K,N] row-major  ->  f16 [Npad,K] (row n = column n of W)
// Pad rows (n >= N) are zero-filled so WMMA B-fragment loads stay clean.
// ---------------------------------------------------------------------------
__global__ __launch_bounds__(256) void prep_w_k(const float* __restrict__ w,
                                                _Float16* __restrict__ wt,
                                                int K, int N, int Npad) {
  size_t idx = (size_t)blockIdx.x * 256 + threadIdx.x;
  if (idx >= (size_t)Npad * K) return;
  int n  = (int)(idx / K);
  int kk = (int)(idx % K);
  wt[idx] = (n < N) ? (_Float16)w[(size_t)kk * N + n] : (_Float16)0.0f;
}

// ---------------------------------------------------------------------------
// Embedding + sinusoidal PE (log(1000) base, as in reference). f32 + f16 out.
// ---------------------------------------------------------------------------
__global__ __launch_bounds__(256) void embed_k(const int* __restrict__ tok,
                                               const float* __restrict__ emb,
                                               float* __restrict__ out32,
                                               _Float16* __restrict__ out16) {
  int idx = blockIdx.x * 256 + threadIdx.x;       // over Mrows*Dm
  int d   = idx & (Dm - 1);
  int row = idx >> 9;                              // b*S + s
  int s   = row & (Ss - 1);
  int t   = tok[row];
  int j2  = (d >> 1) << 1;                         // even base index
  float ang = (float)s * __expf(-(float)j2 * (6.907755278982137f / (float)Dm));
  float pe  = (d & 1) ? __cosf(ang) : __sinf(ang);
  float v   = emb[(size_t)t * Dm + d] + pe;
  out32[idx] = v;
  out16[idx] = (_Float16)v;
}

// ---------------------------------------------------------------------------
// LayerNorm over D=512. One block (128 thr) per row. f32 + f16 out.
// ---------------------------------------------------------------------------
__global__ __launch_bounds__(128) void layernorm_k(const float* __restrict__ in,
                                                   const float* __restrict__ g,
                                                   const float* __restrict__ b,
                                                   float* __restrict__ out32,
                                                   _Float16* __restrict__ out16) {
  __shared__ float s1[128], s2[128];
  int row = blockIdx.x;
  const float* x = in + (size_t)row * Dm;
  float vals[4], lsum = 0.f, lsq = 0.f;
#pragma unroll
  for (int i = 0; i < 4; ++i) {
    float v = x[threadIdx.x + i * 128];
    vals[i] = v; lsum += v; lsq += v * v;
  }
  s1[threadIdx.x] = lsum; s2[threadIdx.x] = lsq;
  __syncthreads();
  for (int off = 64; off > 0; off >>= 1) {
    if (threadIdx.x < off) {
      s1[threadIdx.x] += s1[threadIdx.x + off];
      s2[threadIdx.x] += s2[threadIdx.x + off];
    }
    __syncthreads();
  }
  float mean = s1[0] * (1.0f / Dm);
  float var  = s2[0] * (1.0f / Dm) - mean * mean;
  float inv  = rsqrtf(var + 1e-5f);
#pragma unroll
  for (int i = 0; i < 4; ++i) {
    int c = threadIdx.x + i * 128;
    float o = (vals[i] - mean) * inv * g[c] + b[c];
    out32[(size_t)row * Dm + c] = o;
    out16[(size_t)row * Dm + c] = (_Float16)o;
  }
}

// ---------------------------------------------------------------------------
// V transpose: f16 [B,S,D] -> f16 [B,H,HD,S]  (so P@V B-fragments are contiguous)
// ---------------------------------------------------------------------------
__global__ __launch_bounds__(256) void transpose_v_k(const _Float16* __restrict__ v,
                                                     _Float16* __restrict__ vt) {
  int idx = blockIdx.x * 256 + threadIdx.x;       // over Bb*Ss*Dm
  int d = idx & (Dm - 1);
  int s = (idx >> 9) & (Ss - 1);
  int b = idx >> 18;
  int h = d >> 6, e = d & (HDd - 1);
  vt[(((size_t)(b * Hh + h) * HDd + e) * Ss) + s] = v[idx];
}

// ---------------------------------------------------------------------------
// WMMA GEMM, LDS-staged + double buffered:
//   C[M,N] = A_f16[M,K] * Bt_f16[Npad,K]^T (+bias)(+ReLU)(+resid)
// Block: 256 thr = 8 waves (4x2), block tile 64x128, wave tile 16x64, k-step 32.
// Per k-step the 64x32 A tile and 128x32 B tile are staged to LDS with the
// async global->LDS path (fallback: load+ds_store); fragments come from LDS.
// ---------------------------------------------------------------------------
__global__ __launch_bounds__(256) void gemm_wmma_k(const _Float16* __restrict__ A,
                                                   const _Float16* __restrict__ Bt,
                                                   const float* __restrict__ bias,
                                                   const float* __restrict__ resid,
                                                   float* __restrict__ Cf32,
                                                   _Float16* __restrict__ Cf16,
                                                   int M, int N, int K, int relu) {
  __shared__ __align__(16) _Float16 sA[2][64 * 32];    // [row][k]   4KB x2
  __shared__ __align__(16) _Float16 sB[2][128 * 32];   // [col][k]   8KB x2

  int tid  = threadIdx.x;
  int wave = tid >> 5;
  int lane = tid & 31;
  int l15  = lane & 15;
  int hs   = lane >> 4;                       // half-wave select (K half: k = 16*hs + s)
  int wrow = wave >> 1, wcol = wave & 1;
  int bm0  = blockIdx.y * 64;
  int bn0  = blockIdx.x * 128;

  // staging assignments (per thread): A: 1x16B chunk, B: 2x16B chunks
  int arow = tid >> 2, achk = (tid & 3) * 8;
  const _Float16* agp = A  + (size_t)(bm0 + arow) * K + achk;

  v8f acc[4] = {};

  // ---- prologue: stage k0 = 0 into buffer 0 ----
  {
    async_cp16(agp, &sA[0][arow * 32 + achk]);
#pragma unroll
    for (int j = 0; j < 2; ++j) {
      int idx = tid + j * 256;
      int br = idx >> 2, bc = (idx & 3) * 8;
      async_cp16(Bt + (size_t)(bn0 + br) * K + bc, &sB[0][br * 32 + bc]);
    }
  }
  async_drain();
  __syncthreads();

  int kb = 0;
  for (int k0 = 0; k0 < K; k0 += 32) {
    // ---- stage next k-step while computing this one ----
    if (k0 + 32 < K) {
      int kn = k0 + 32;
      if (kn + 96 < K) __builtin_prefetch(agp + kn + 96, 0, 0);
      async_cp16(agp + kn, &sA[kb ^ 1][arow * 32 + achk]);
#pragma unroll
      for (int j = 0; j < 2; ++j) {
        int idx = tid + j * 256;
        int br = idx >> 2, bc = (idx & 3) * 8;
        async_cp16(Bt + (size_t)(bn0 + br) * K + kn + bc,
                   &sB[kb ^ 1][br * 32 + bc]);
      }
    }
    // ---- fragments from LDS, 4 WMMAs ----
    const _Float16* sa = &sA[kb][(wrow * 16 + l15) * 32 + hs * 16];
    AFrag a;
    a.h[0] = *(const v8h*)(sa);
    a.h[1] = *(const v8h*)(sa + 8);
#pragma unroll
    for (int nt = 0; nt < 4; ++nt) {
      const _Float16* sb = &sB[kb][(wcol * 64 + nt * 16 + l15) * 32 + hs * 16];
      AFrag b;
      b.h[0] = *(const v8h*)(sb);
      b.h[1] = *(const v8h*)(sb + 8);
      acc[nt] = wmma_acc(a, b, acc[nt]);
    }
    async_drain();
    __syncthreads();
    kb ^= 1;
  }

  // ---- epilogue: bias / relu / residual, f32 and/or f16 stores ----
  int rowbase = bm0 + wrow * 16 + hs * 8;
#pragma unroll
  for (int nt = 0; nt < 4; ++nt) {
    int col = bn0 + wcol * 64 + nt * 16 + l15;
    if (col >= N) continue;
    float bv = bias ? bias[col] : 0.0f;
#pragma unroll
    for (int r = 0; r < 8; ++r) {
      int row = rowbase + r;
      float v = acc[nt][r] + bv;
      if (relu) v = fmaxf(v, 0.0f);
      if (resid) v += resid[(size_t)row * N + col];
      if (Cf32) Cf32[(size_t)row * N + col] = v;
      if (Cf16) Cf16[(size_t)row * N + col] = (_Float16)v;
    }
  }
}

// ---------------------------------------------------------------------------
// Fused attention: one block per (b,h,16-query tile). Scores + softmax + P@V.
// q,k: f16 [B,Lq/Lk,D] ; vt: f16 [B,H,HD,Lk] ; out: f16 [B,Lq,D]
// ---------------------------------------------------------------------------
__global__ __launch_bounds__(256) void attn_k(const _Float16* __restrict__ qb,
                                              const _Float16* __restrict__ kb,
                                              const _Float16* __restrict__ vt,
                                              _Float16* __restrict__ ob,
                                              int Lq, int Lk, int causal) {
  __shared__ __align__(16) float    sS[16 * 512];
  __shared__ __align__(16) _Float16 sP[16 * 512];
  __shared__ __align__(16) float    sO[16 * 64];
  __shared__ float sRed[16 * 16];
  __shared__ float sInv[16];

  int b = blockIdx.z, h = blockIdx.y;
  int q0 = blockIdx.x * 16;
  int tid = threadIdx.x, wave = tid >> 5, lane = tid & 31;
  int l15 = lane & 15, hs = lane >> 4;

  // --- Q fragments (16 rows x K=64, two k-steps of 32) ---
  const _Float16* qrow = qb + ((size_t)(b * Lq + q0 + l15) * Dm) + h * HDd;
  AFrag aq0, aq1;
  aq0.h[0] = *(const v8h*)(qrow + hs * 16);
  aq0.h[1] = *(const v8h*)(qrow + hs * 16 + 8);
  aq1.h[0] = *(const v8h*)(qrow + 32 + hs * 16);
  aq1.h[1] = *(const v8h*)(qrow + 32 + hs * 16 + 8);

  // --- scores: wave owns 4 key tiles -> 64 columns of the 16x512 strip ---
#pragma unroll
  for (int t = 0; t < 4; ++t) {
    int kt  = wave * 4 + t;
    int key = kt * 16 + l15;
    const _Float16* krow = kb + ((size_t)(b * Lk + key) * Dm) + h * HDd;
    AFrag b0, b1;
    b0.h[0] = *(const v8h*)(krow + hs * 16);
    b0.h[1] = *(const v8h*)(krow + hs * 16 + 8);
    b1.h[0] = *(const v8h*)(krow + 32 + hs * 16);
    b1.h[1] = *(const v8h*)(krow + 32 + hs * 16 + 8);
    v8f acc = {};
    acc = wmma_acc(aq0, b0, acc);
    acc = wmma_acc(aq1, b1, acc);
#pragma unroll
    for (int r = 0; r < 8; ++r)
      sS[(hs * 8 + r) * 512 + kt * 16 + l15] = acc[r] * 0.125f;  // 1/sqrt(64)
  }
  __syncthreads();

  // --- softmax: 16 threads per row, 32 cols each ---
  int row = tid & 15, sub = tid >> 4;
  int qglob = q0 + row;
  float mx = -3.0e38f;
  for (int j = 0; j < 32; ++j) {
    int c = sub * 32 + j;
    if (!(causal && c > qglob)) mx = fmaxf(mx, sS[row * 512 + c]);
  }
  sRed[row * 16 + sub] = mx;
  __syncthreads();
  float gmx = -3.0e38f;
  for (int j = 0; j < 16; ++j) gmx = fmaxf(gmx, sRed[row * 16 + j]);
  __syncthreads();
  float lsum = 0.f;
  for (int j = 0; j < 32; ++j) {
    int c = sub * 32 + j;
    float p = (causal && c > qglob) ? 0.0f : __expf(sS[row * 512 + c] - gmx);
    sP[row * 512 + c] = (_Float16)p;
    lsum += p;
  }
  sRed[row * 16 + sub] = lsum;
  __syncthreads();
  float gs = 0.f;
  for (int j = 0; j < 16; ++j) gs += sRed[row * 16 + j];
  if (sub == 0) sInv[row] = 1.0f / gs;
  for (int i = tid; i < 16 * 64; i += 256) sO[i] = 0.0f;
  __syncthreads();

  // --- O = P @ V : wave owns keys [wave*64, wave*64+64) (2 k-steps) ---
  v8f oacc[4] = {};
  const _Float16* vbase = vt + (size_t)(b * Hh + h) * HDd * (size_t)Lk;
#pragma unroll
  for (int s = 0; s < 2; ++s) {
    int k0 = wave * 64 + s * 32;
    AFrag ap;
    const _Float16* prow = &sP[l15 * 512 + k0 + hs * 16];
    ap.h[0] = *(const v8h*)(prow);
    ap.h[1] = *(const v8h*)(prow + 8);
#pragma unroll
    for (int nt = 0; nt < 4; ++nt) {
      const _Float16* vrow = vbase + (size_t)(nt * 16 + l15) * Lk + k0 + hs * 16;
      AFrag bv;
      bv.h[0] = *(const v8h*)(vrow);
      bv.h[1] = *(const v8h*)(vrow + 8);
      oacc[nt] = wmma_acc(ap, bv, oacc[nt]);
    }
  }
#pragma unroll
  for (int nt = 0; nt < 4; ++nt)
#pragma unroll
    for (int r = 0; r < 8; ++r)
      atomicAdd(&sO[(hs * 8 + r) * 64 + nt * 16 + l15], oacc[nt][r]);
  __syncthreads();

  // --- write [16,64] tile into [B,Lq,D] layout (heads already merged) ---
  for (int i = tid; i < 16 * 64; i += 256) {
    int r = i >> 6, c = i & 63;
    ob[((size_t)(b * Lq + q0 + r) * Dm) + h * HDd + c] = (_Float16)(sO[i] * sInv[r]);
  }
}

// ---------------------------------------------------------------------------
// Host driver
// ---------------------------------------------------------------------------
extern "C" void kernel_launch(void* const* d_in, const int* in_sizes, int n_in,
                              void* d_out, int out_size, void* d_ws, size_t ws_size,
                              hipStream_t stream) {
  (void)in_sizes; (void)n_in; (void)out_size; (void)ws_size;

  const int*   src    = (const int*)d_in[0];
  const int*   tgt    = (const int*)d_in[1];
  const float* emb    = (const float*)d_in[2];
  const float* enc_wq = (const float*)d_in[3];
  const float* enc_wk = (const float*)d_in[4];
  const float* enc_wv = (const float*)d_in[5];
  const float* enc_wo = (const float*)d_in[6];
  const float* enc_up = (const float*)d_in[7];
  const float* enc_dn = (const float*)d_in[8];
  const float* enc_bq = (const float*)d_in[9];
  const float* enc_bk = (const float*)d_in[10];
  const float* enc_bv = (const float*)d_in[11];
  const float* enc_bo = (const float*)d_in[12];
  const float* enc_ub = (const float*)d_in[13];
  const float* enc_db = (const float*)d_in[14];
  const float* enc_lb = (const float*)d_in[15];
  const float* enc_lg = (const float*)d_in[16];
  const float* dec_wq = (const float*)d_in[17];
  const float* dec_wk = (const float*)d_in[18];
  const float* dec_wv = (const float*)d_in[19];
  const float* dec_wo = (const float*)d_in[20];
  const float* dec_up = (const float*)d_in[21];
  const float* dec_dn = (const float*)d_in[22];
  const float* dec_bq = (const float*)d_in[23];
  const float* dec_bk = (const float*)d_in[24];
  const float* dec_bv = (const float*)d_in[25];
  const float* dec_bo = (const float*)d_in[26];
  const float* dec_ub = (const float*)d_in[27];
  const float* dec_db = (const float*)d_in[28];
  const float* dec_lb = (const float*)d_in[29];
  const float* dec_lg = (const float*)d_in[30];
  const float* out_w  = (const float*)d_in[31];
  const float* out_b  = (const float*)d_in[32];

  // ---- workspace carve-out (~190 MB total) ----
  char* ws = (char*)d_ws;
  size_t off = 0;
  auto alloc = [&](size_t bytes) -> char* {
    char* p = ws + off;
    off += (bytes + 255) & ~(size_t)255;
    return p;
  };
  auto h16buf = [&](size_t n) { return (_Float16*)alloc(n * sizeof(_Float16)); };
  auto f32buf = [&](size_t n) { return (float*)alloc(n * sizeof(float)); };

  // transposed f16 weights
  _Float16* ewq_t = h16buf((size_t)Ll * DDsz);
  _Float16* ewk_t = h16buf((size_t)Ll * DDsz);
  _Float16* ewv_t = h16buf((size_t)Ll * DDsz);
  _Float16* ewo_t = h16buf((size_t)Ll * DDsz);
  _Float16* eup_t = h16buf((size_t)Ll * DFFsz);
  _Float16* edn_t = h16buf((size_t)Ll * DFFsz);
  _Float16* dwq_t = h16buf((size_t)Ll * DDsz);
  _Float16* dwk_t = h16buf((size_t)Ll * DDsz);
  _Float16* dwv_t = h16buf((size_t)Ll * DDsz);
  _Float16* dwo_t = h16buf((size_t)Ll * DDsz);
  _Float16* dup_t = h16buf((size_t)Ll * DFFsz);
  _Float16* ddn_t = h16buf((size_t)Ll * DFFsz);
  _Float16* owt   = h16buf((size_t)VPAD * Dm);

  // activations
  const size_t MD = (size_t)Mrows * Dm;
  float*    xf   = f32buf(MD);           // encoder activation f32
  float*    yf   = f32buf(MD);           // decoder activation f32
  float*    r1f  = f32buf(MD);
  float*    r2f  = f32buf(MD);
  float*    fif  = f32buf(MD);
  float*    cif  = f32buf(MD);
  _Float16* x16  = h16buf(MD);           // encoder activation f16 (== memory f16 after enc)
  _Float16* y16  = h16buf(MD);
  _Float16* q16  = h16buf(MD);
  _Float16* k16  = h16buf(MD);
  _Float16* v16b = h16buf(MD);
  _Float16* vt16 = h16buf(MD);
  _Float16* at16 = h16buf(MD);
  _Float16* fi16 = h16buf(MD);
  _Float16* ci16 = h16buf(MD);
  _Float16* h16  = h16buf((size_t)Mrows * FFf);

  // ---- launch helpers ----
  auto prep = [&](const float* w, _Float16* wt, int K, int N, int Npad) {
    size_t tot = (size_t)Npad * K;
    prep_w_k<<<(unsigned)((tot + 255) / 256), 256, 0, stream>>>(w, wt, K, N, Npad);
  };
  auto gemm = [&](const _Float16* A, const _Float16* Bt, const float* bias,
                  const float* resid, float* c32, _Float16* c16,
                  int N, int K, int relu) {
    dim3 g((N + 127) / 128, Mrows / 64);
    gemm_wmma_k<<<g, 256, 0, stream>>>(A, Bt, bias, resid, c32, c16,
                                       Mrows, N, K, relu);
  };
  auto attn = [&](const _Float16* q, const _Float16* k, const _Float16* vtb,
                  _Float16* o, int causal) {
    dim3 g(Ss / 16, Hh, Bb);
    attn_k<<<g, 256, 0, stream>>>(q, k, vtb, o, Ss, Ss, causal);
  };
  auto transp = [&](const _Float16* v, _Float16* vtb) {
    transpose_v_k<<<(Bb * Ss * Dm) / 256, 256, 0, stream>>>(v, vtb);
  };
  auto ln = [&](const float* in, const float* g, const float* b,
                float* o32, _Float16* o16) {
    layernorm_k<<<Mrows, 128, 0, stream>>>(in, g, b, o32, o16);
  };

  // ---- one-time (per call) weight conversion/transpose ----
  for (int i = 0; i < Ll; ++i) {
    prep(enc_wq + (size_t)i * DDsz,  ewq_t + (size_t)i * DDsz,  Dm, Dm, Dm);
    prep(enc_wk + (size_t)i * DDsz,  ewk_t + (size_t)i * DDsz,  Dm, Dm, Dm);
    prep(enc_wv + (size_t)i * DDsz,  ewv_t + (size_t)i * DDsz,  Dm, Dm, Dm);
    prep(enc_wo + (size_t)i * DDsz,  ewo_t + (size_t)i * DDsz,  Dm, Dm, Dm);
    prep(enc_up + (size_t)i * DFFsz, eup_t + (size_t)i * DFFsz, Dm, FFf, FFf);
    prep(enc_dn + (size_t)i * DFFsz, edn_t + (size_t)i * DFFsz, FFf, Dm, Dm);
    prep(dec_wq + (size_t)i * DDsz,  dwq_t + (size_t)i * DDsz,  Dm, Dm, Dm);
    prep(dec_wk + (size_t)i * DDsz,  dwk_t + (size_t)i * DDsz,  Dm, Dm, Dm);
    prep(dec_wv + (size_t)i * DDsz,  dwv_t + (size_t)i * DDsz,  Dm, Dm, Dm);
    prep(dec_wo + (size_t)i * DDsz,  dwo_t + (size_t)i * DDsz,  Dm, Dm, Dm);
    prep(dec_up + (size_t)i * DFFsz, dup_t + (size_t)i * DFFsz, Dm, FFf, FFf);
    prep(dec_dn + (size_t)i * DFFsz, ddn_t + (size_t)i * DFFsz, FFf, Dm, Dm);
  }
  prep(out_w, owt, Dm, Vv, VPAD);

  // ---- embeddings + PE ----
  embed_k<<<(Mrows * Dm) / 256, 256, 0, stream>>>(src, emb, xf, x16);
  embed_k<<<(Mrows * Dm) / 256, 256, 0, stream>>>(tgt, emb, yf, y16);

  // ---- encoder stack ----
  for (int i = 0; i < Ll; ++i) {
    const _Float16 *wq = ewq_t + (size_t)i * DDsz, *wk = ewk_t + (size_t)i * DDsz,
                   *wv = ewv_t + (size_t)i * DDsz, *wo = ewo_t + (size_t)i * DDsz,
                   *up = eup_t + (size_t)i * DFFsz, *dn = edn_t + (size_t)i * DFFsz;
    const float *bq = enc_bq + i * Dm, *bk = enc_bk + i * Dm, *bv = enc_bv + i * Dm,
                *bo = enc_bo + i * Dm, *ub = enc_ub + i * FFf, *db = enc_db + i * Dm,
                *lg = enc_lg + i * Dm, *lb = enc_lb + i * Dm;
    gemm(x16, wq, bq, nullptr, nullptr, q16, Dm, Dm, 0);
    gemm(x16, wk, bk, nullptr, nullptr, k16, Dm, Dm, 0);
    gemm(x16, wv, bv, nullptr, nullptr, v16b, Dm, Dm, 0);
    transp(v16b, vt16);
    attn(q16, k16, vt16, at16, /*causal=*/0);
    gemm(at16, wo, bo, /*resid=*/xf, r1f, nullptr, Dm, Dm, 0);   // x1 = x + a
    ln(r1f, lg, lb, fif, fi16);                                  // fi
    gemm(fi16, up, ub, nullptr, nullptr, h16, FFf, Dm, /*relu=*/1);
    gemm(h16, dn, db, /*resid=*/fif, r2f, nullptr, Dm, FFf, 0);  // fo
    ln(r2f, lg, lb, xf, x16);                                    // x = LN(fo)
  }
  // memory (f16) == x16

  // ---- decoder stack ----
  for (int i = 0; i < Ll; ++i) {
    const _Float16 *wq = dwq_t + (size_t)i * DDsz, *wk = dwk_t + (size_t)i * DDsz,
                   *wv = dwv_t + (size_t)i * DDsz, *wo = dwo_t + (size_t)i * DDsz,
                   *up = dup_t + (size_t)i * DFFsz, *dn = ddn_t + (size_t)i * DFFsz;
    const float *bq = dec_bq + i * Dm, *bk = dec_bk + i * Dm, *bv = dec_bv + i * Dm,
                *bo = dec_bo + i * Dm, *ub = dec_ub + i * FFf, *db = dec_db + i * Dm,
                *lg = dec_lg + i * Dm, *lb = dec_lb + i * Dm;
    // causal self-attention
    gemm(y16, wq, bq, nullptr, nullptr, q16, Dm, Dm, 0);
    gemm(y16, wk, bk, nullptr, nullptr, k16, Dm, Dm, 0);
    gemm(y16, wv, bv, nullptr, nullptr, v16b, Dm, Dm, 0);
    transp(v16b, vt16);
    attn(q16, k16, vt16, at16, /*causal=*/1);
    gemm(at16, wo, bo, /*resid=*/yf, r1f, nullptr, Dm, Dm, 0);   // y1 = y + sa
    ln(r1f, lg, lb, cif, ci16);                                  // ci
    // cross-attention (same weights, keys/values from encoder memory x16)
    gemm(ci16, wq, bq, nullptr, nullptr, q16, Dm, Dm, 0);
    gemm(x16, wk, bk, nullptr, nullptr, k16, Dm, Dm, 0);
    gemm(x16, wv, bv, nullptr, nullptr, v16b, Dm, Dm, 0);
    transp(v16b, vt16);
    attn(q16, k16, vt16, at16, /*causal=*/0);
    gemm(at16, wo, bo, /*resid=*/cif, r2f, nullptr, Dm, Dm, 0);  // y2 = ci + ca
    ln(r2f, lg, lb, fif, fi16);                                  // fi
    gemm(fi16, up, ub, nullptr, nullptr, h16, FFf, Dm, /*relu=*/1);
    gemm(h16, dn, db, /*resid=*/fif, r1f, nullptr, Dm, FFf, 0);  // fo
    ln(r1f, lg, lb, yf, y16);                                    // y = LN(fo)
  }

  // ---- output projection: logits [B*T, V] f32 ----
  gemm(y16, owt, out_b, nullptr, (float*)d_out, nullptr, Vv, Dm, 0);
}
